// MonotonicEnergy_22411139350722
// MI455X (gfx1250) — matrix operand order repaired and checked by reading, not compiled, not run on hardware
//
#include <hip/hip_runtime.h>
#include <stdint.h>

// ---------------------------------------------------------------------------
// MonotonicEnergy on gfx1250 (MI455X).
// GEMMs on V_WMMA_F32_16X16X32_BF16 with bf16x3 split (x = hi + lo):
//   A.B ~= Ah.Bh + Ah.Bl + Al.Bh   -> ~2^-16 rel error at bf16 throughput.
// hi/lo conversion happens exactly once per element (proj epilogue + W
// pre-pass). Tile staging uses CDNA5 async global->LDS b128 DMA (ASYNCcnt)
// with ping-pong LDS buffers: slab s+1 is issued before computing slab s and
// drained after it, one barrier per iteration (phases touch disjoint buffers).
// ---------------------------------------------------------------------------

typedef __attribute__((ext_vector_type(16))) __bf16 v16bf;
typedef __attribute__((ext_vector_type(8)))  __bf16 v8bf;
typedef __attribute__((ext_vector_type(8)))  float  v8f;

#define B_    16
#define KLEN_ 2048
#define QLEN_ 512
#define ADIM_ 512
#define H_    4
#define DK_   128
#define LSTR  40   // LDS row stride (bf16 elems): 80B, 16B-aligned, bank-spread

// ---- CDNA5 async global->LDS copy (16B), tracked by ASYNCcnt ---------------
__device__ __forceinline__ void async_cp16(const void* g, void* l) {
    unsigned lofs = (unsigned)(uintptr_t)l;               // low 32b = LDS addr
    unsigned long long ga = (unsigned long long)(uintptr_t)g;
    asm volatile("global_load_async_to_lds_b128 %0, %1, off"
                 :: "v"(lofs), "v"(ga) : "memory");
}
#define WAIT_ASYNC0() asm volatile("s_wait_asynccnt 0" ::: "memory")

__device__ __forceinline__ void split8(const float* xs, v8bf& h, v8bf& l) {
    #pragma unroll
    for (int i = 0; i < 8; ++i) {
        __bf16 hb = (__bf16)xs[i];
        h[i] = hb;
        l[i] = (__bf16)(xs[i] - (float)hb);
    }
}

// A-frag (16x32 bf16): lane half lh -> K chunks [8lh,8lh+8) and [16+8lh,+8)
__device__ __forceinline__ v16bf load_afrag(const __bf16* row, int lh) {
    v16bf f;
    *(v8bf*)&f       = *(const v8bf*)(row + 8 * lh);
    *((v8bf*)&f + 1) = *(const v8bf*)(row + 16 + 8 * lh);
    return f;
}

// B-frag (32x16 bf16): lane half lh -> K chunks [8lh,8lh+8) and [16+8lh,+8)
__device__ __forceinline__ v16bf load_bfrag(const __bf16* row, int lh) {
    v16bf f;
    *(v8bf*)&f       = *(const v8bf*)(row + 8 * lh);
    *((v8bf*)&f + 1) = *(const v8bf*)(row + 16 + 8 * lh);
    return f;
}

#define WMMA_BF16(A, Bm, C) \
    C = __builtin_amdgcn_wmma_f32_16x16x32_bf16(false, A, false, Bm, (short)0, C, false, false)

// ---- one-time W pre-pass: W[K][N] f32 -> transposed bf16 hi/lo [N][K] ------
__global__ __launch_bounds__(256) void wpre_kernel(
    const float* __restrict__ W, __bf16* __restrict__ Th, __bf16* __restrict__ Tl)
{
    const int idx = blockIdx.x * 256 + threadIdx.x;   // 0 .. 512*512-1
    const int k = idx >> 9;
    const int n = idx & 511;
    const float x = W[idx];                           // W[k*512 + n]
    const __bf16 hb = (__bf16)x;
    Th[n * 512 + k] = hb;
    Tl[n * 512 + k] = (__bf16)(x - (float)hb);
}

// -------------------- projection GEMM -> bf16 hi/lo planes ------------------
// C[M,N] = X[M,K] @ W[K,N] + bias, emitted as Oh/Ol bf16 planes.
// block tile 64(M) x 128(N); 8 waves in 2x4; wave = 2x2 WMMA tiles.
// Double-buffered: W via async DMA, X prefetched to regs, converted post-WMMA.
__global__ __launch_bounds__(256) void proj_kernel(
    const float* __restrict__ X,
    const __bf16* __restrict__ WthG, const __bf16* __restrict__ WtlG,
    const float* __restrict__ bias,
    __bf16* __restrict__ Oh, __bf16* __restrict__ Ol,
    int M, int Kdim, int N)
{
    __shared__ __bf16 Xh[2][64 * LSTR],   Xl[2][64 * LSTR];
    __shared__ __bf16 Wth[2][128 * LSTR], Wtl[2][128 * LSTR];

    const int t    = threadIdx.x;
    const int lane = t & 31;
    const int wave = t >> 5;
    const int wm   = wave & 1;
    const int wn   = wave >> 1;
    const int l16  = lane & 15;
    const int lh   = lane >> 4;
    const int m0   = blockIdx.y * 64;
    const int n0   = blockIdx.x * 128;

    const int xr  = t >> 2;          // X row 0..63
    const int xc  = (t & 3) * 8;     // X col group 0,8,16,24
    const int wnl = t & 127;         // W transposed row (n)
    const int wkh = t >> 7;          // W k half (0/1)

    v8f acc[2][2] = {};

    auto issueW = [&](int k0, int pb) {
        const size_t ws = (size_t)(n0 + wnl) * Kdim + k0 + 16 * wkh;
        __bf16* dh = &Wth[pb][wnl * LSTR + 16 * wkh];
        __bf16* dl = &Wtl[pb][wnl * LSTR + 16 * wkh];
        async_cp16(&WthG[ws],     dh);
        async_cp16(&WthG[ws + 8], dh + 8);
        async_cp16(&WtlG[ws],     dl);
        async_cp16(&WtlG[ws + 8], dl + 8);
    };
    auto loadX = [&](int k0, float* xs) {
        const float* xp = &X[(size_t)(m0 + xr) * Kdim + k0 + xc];
        float4 u = *(const float4*)xp;
        float4 v = *(const float4*)(xp + 4);
        xs[0] = u.x; xs[1] = u.y; xs[2] = u.z; xs[3] = u.w;
        xs[4] = v.x; xs[5] = v.y; xs[6] = v.z; xs[7] = v.w;
    };
    auto commitX = [&](const float* xs, int pb) {
        v8bf h, l;
        split8(xs, h, l);
        *(v8bf*)&Xh[pb][xr * LSTR + xc] = h;
        *(v8bf*)&Xl[pb][xr * LSTR + xc] = l;
    };
    auto compute = [&](int pb) {
        v16bf ah[2], al[2], bh[2], bl[2];
        #pragma unroll
        for (int mt = 0; mt < 2; ++mt) {
            ah[mt] = load_afrag(&Xh[pb][(wm * 32 + mt * 16 + l16) * LSTR], lh);
            al[mt] = load_afrag(&Xl[pb][(wm * 32 + mt * 16 + l16) * LSTR], lh);
        }
        #pragma unroll
        for (int nt = 0; nt < 2; ++nt) {
            bh[nt] = load_bfrag(&Wth[pb][(wn * 32 + nt * 16 + l16) * LSTR], lh);
            bl[nt] = load_bfrag(&Wtl[pb][(wn * 32 + nt * 16 + l16) * LSTR], lh);
        }
        #pragma unroll
        for (int mt = 0; mt < 2; ++mt)
            #pragma unroll
            for (int nt = 0; nt < 2; ++nt) {
                WMMA_BF16(ah[mt], bh[nt], acc[mt][nt]);
                WMMA_BF16(ah[mt], bl[nt], acc[mt][nt]);
                WMMA_BF16(al[mt], bh[nt], acc[mt][nt]);
            }
    };

    const int NS = Kdim >> 5;   // 32-wide K slabs

    // prologue: stage slab 0 into buffer 0
    issueW(0, 0);
    {
        float xs[8];
        loadX(0, xs);
        commitX(xs, 0);
    }
    WAIT_ASYNC0();
    __syncthreads();

    for (int s = 0; s < NS; ++s) {
        const int pb = s & 1;
        const bool more = (s + 1 < NS);
        float xn[8];
        if (more) {                      // issue next slab before computing
            issueW((s + 1) << 5, pb ^ 1);
            loadX((s + 1) << 5, xn);
        }
        compute(pb);
        if (more) commitX(xn, pb ^ 1);   // cvt+store after WMMA block
        WAIT_ASYNC0();                   // drain next-slab DMA (overlapped)
        __syncthreads();                 // single barrier: phases ping-pong
    }

    // epilogue: +bias, split once to bf16 hi/lo planes
    #pragma unroll
    for (int mt = 0; mt < 2; ++mt) {
        #pragma unroll
        for (int nt = 0; nt < 2; ++nt) {
            const int col = n0 + wn * 32 + nt * 16 + l16;
            const float bv = bias[col];
            #pragma unroll
            for (int rr = 0; rr < 8; ++rr) {
                const int row = m0 + wm * 32 + mt * 16 + rr + 8 * lh;
                const float val = acc[mt][nt][rr] + bv;
                const __bf16 hb = (__bf16)val;
                const size_t o = (size_t)row * N + col;
                Oh[o] = hb;
                Ol[o] = (__bf16)(val - (float)hb);
            }
        }
    }
}

// -------------------- energy: per (b,h) GEMM  E = Qp . Kp^T  ----------------
// block tile 64(q) x 128(k); staging entirely via async DMA, double-buffered.
__global__ __launch_bounds__(256) void energy_kernel(
    const __bf16* __restrict__ Qhp, const __bf16* __restrict__ Qlp,
    const __bf16* __restrict__ Khp, const __bf16* __restrict__ Klp,
    const int* __restrict__ mask, const float* __restrict__ rptr,
    float* __restrict__ out)
{
    __shared__ __bf16 Qh[2][64 * LSTR],  Ql[2][64 * LSTR];
    __shared__ __bf16 Kh[2][128 * LSTR], Kl[2][128 * LSTR];

    const int t    = threadIdx.x;
    const int lane = t & 31;
    const int wave = t >> 5;
    const int wm   = wave & 1;
    const int wn   = wave >> 1;
    const int l16  = lane & 15;
    const int lh   = lane >> 4;
    const int bz   = blockIdx.z;          // b*H + h
    const int b    = bz >> 2;
    const int h    = bz & 3;
    const int i0   = blockIdx.y * 64;
    const int j0   = blockIdx.x * 128;

    const float rval      = rptr[0];
    const float inv_scale = 0.04419417382415922f;   // 1/sqrt(512)

    const int qr = t >> 2;            // 0..63
    const int qc = (t & 3) * 8;       // 0,8,16,24
    const int kr = t >> 1;            // 0..127
    const int kc = (t & 1) * 16;      // 0,16

    v8f acc[2][2] = {};

    auto issueQK = [&](int d0, int pb) {
        const size_t qs = ((size_t)(b * QLEN_ + i0 + qr)) * ADIM_ + h * DK_ + d0 + qc;
        async_cp16(&Qhp[qs], &Qh[pb][qr * LSTR + qc]);
        async_cp16(&Qlp[qs], &Ql[pb][qr * LSTR + qc]);
        const size_t ks = ((size_t)(b * KLEN_ + j0 + kr)) * ADIM_ + h * DK_ + d0 + kc;
        __bf16* dh = &Kh[pb][kr * LSTR + kc];
        __bf16* dl = &Kl[pb][kr * LSTR + kc];
        async_cp16(&Khp[ks],     dh);
        async_cp16(&Khp[ks + 8], dh + 8);
        async_cp16(&Klp[ks],     dl);
        async_cp16(&Klp[ks + 8], dl + 8);
    };
    auto compute = [&](int pb) {
        v16bf ah[2], al[2], bh[2], bl[2];
        #pragma unroll
        for (int mt = 0; mt < 2; ++mt) {
            ah[mt] = load_afrag(&Qh[pb][(wm * 32 + mt * 16 + l16) * LSTR], lh);
            al[mt] = load_afrag(&Ql[pb][(wm * 32 + mt * 16 + l16) * LSTR], lh);
        }
        #pragma unroll
        for (int nt = 0; nt < 2; ++nt) {
            bh[nt] = load_bfrag(&Kh[pb][(wn * 32 + nt * 16 + l16) * LSTR], lh);
            bl[nt] = load_bfrag(&Kl[pb][(wn * 32 + nt * 16 + l16) * LSTR], lh);
        }
        #pragma unroll
        for (int mt = 0; mt < 2; ++mt)
            #pragma unroll
            for (int nt = 0; nt < 2; ++nt) {
                WMMA_BF16(ah[mt], bh[nt], acc[mt][nt]);
                WMMA_BF16(ah[mt], bl[nt], acc[mt][nt]);
                WMMA_BF16(al[mt], bh[nt], acc[mt][nt]);
            }
    };

    // prologue
    issueQK(0, 0);
    WAIT_ASYNC0();
    __syncthreads();

    #pragma unroll
    for (int s = 0; s < (DK_ / 32); ++s) {
        const int pb = s & 1;
        if (s + 1 < (DK_ / 32)) issueQK((s + 1) * 32, pb ^ 1);
        compute(pb);
        WAIT_ASYNC0();
        __syncthreads();
    }

    // epilogue: scale + r, mask -> -FLT_MAX, write [B,H,QLEN,KLEN]
    #pragma unroll
    for (int mt = 0; mt < 2; ++mt) {
        #pragma unroll
        for (int nt = 0; nt < 2; ++nt) {
            const int j = j0 + wn * 32 + nt * 16 + l16;
            #pragma unroll
            for (int rr = 0; rr < 8; ++rr) {
                const int i = i0 + wm * 32 + mt * 16 + rr + 8 * lh;
                const size_t midx = ((size_t)(b * QLEN_ + i)) * KLEN_ + j;
                const int mval = mask[midx];
                const float e = acc[mt][nt][rr] * inv_scale + rval;
                const float v = (mval == 0) ? -3.4028234663852886e+38f : e;
                out[(((size_t)bz) * QLEN_ + i) * KLEN_ + j] = v;
            }
        }
    }
}

extern "C" void kernel_launch(void* const* d_in, const int* in_sizes, int n_in,
                              void* d_out, int out_size, void* d_ws, size_t ws_size,
                              hipStream_t stream) {
    const float* key_in = (const float*)d_in[0];
    const float* query  = (const float*)d_in[1];
    const int*   mask   = (const int*)d_in[2];
    const float* Wk     = (const float*)d_in[3];
    const float* bk     = (const float*)d_in[4];
    const float* Wq     = (const float*)d_in[5];
    const float* bq     = (const float*)d_in[6];
    const float* r      = (const float*)d_in[7];
    float* out = (float*)d_out;

    // workspace layout (bf16 planes), ~82 MB total
    __bf16* Khp  = (__bf16*)d_ws;                              // [32768, 512]
    __bf16* Klp  = Khp  + (size_t)B_ * KLEN_ * ADIM_;
    __bf16* Qhp  = Klp  + (size_t)B_ * KLEN_ * ADIM_;          // [8192, 512]
    __bf16* Qlp  = Qhp  + (size_t)B_ * QLEN_ * ADIM_;
    __bf16* Wkth = Qlp  + (size_t)B_ * QLEN_ * ADIM_;          // [512, 512] each
    __bf16* Wktl = Wkth + (size_t)512 * 512;
    __bf16* Wqth = Wktl + (size_t)512 * 512;
    __bf16* Wqtl = Wqth + (size_t)512 * 512;

    dim3 blk(256);

    // one-time W transpose + bf16 hi/lo split
    wpre_kernel<<<dim3((512 * 512) / 256), blk, 0, stream>>>(Wk, Wkth, Wktl);
    wpre_kernel<<<dim3((512 * 512) / 256), blk, 0, stream>>>(Wq, Wqth, Wqtl);

    // K projection: M = B*KLEN = 32768
    proj_kernel<<<dim3(ADIM_ / 128, (B_ * KLEN_) / 64), blk, 0, stream>>>(
        key_in, Wkth, Wktl, bk, Khp, Klp, B_ * KLEN_, 512, ADIM_);

    // Q projection: M = B*QLEN = 8192
    proj_kernel<<<dim3(ADIM_ / 128, (B_ * QLEN_) / 64), blk, 0, stream>>>(
        query, Wqth, Wqtl, bq, Qhp, Qlp, B_ * QLEN_, 512, ADIM_);

    // energy: grid = (klen tiles, qlen tiles, B*H)
    energy_kernel<<<dim3(KLEN_ / 128, QLEN_ / 64, B_ * H_), blk, 0, stream>>>(
        Qhp, Qlp, Khp, Klp, mask, r, out);
}
